// ChannelCrissCrossAttention_71073118814813
// MI455X (gfx1250) — matrix operand order repaired and checked by reading, not compiled, not run on hardware
//
#include <hip/hip_runtime.h>

// ---------------------------------------------------------------------------
// ChannelCrissCrossAttention on MI455X (gfx1250, wave32, WMMA bf16 16x16x32)
//
// One generic strided batched GEMM kernel, fully specialized at compile time
// on (A-loader shape, B-loader shape, A dtype, B dtype, epilogue mode), plus
// a joint softmax kernel.  bf16 operands, f32 accumulation.
// ---------------------------------------------------------------------------

typedef __attribute__((ext_vector_type(16))) __bf16 v16bf;
typedef __attribute__((ext_vector_type(8)))  float  v8f;

#define PITCH 40   // 32 + 8 pad (u16); 80B rows: 16B-aligned, conflict-free frags

enum { LD_K = 0, LD_MN = 1, LD_NONE = 2 };           // which index is contiguous
enum {
  MODE_STORE_F32    = 0,   // C  = acc
  MODE_ACCUM_F32    = 1,   // C += acc
  MODE_STORE_BF16_B = 2,   // C  = bf16(acc + bias[m])
  MODE_FINAL        = 3    // C  = gamma*(acc + 6*bias[m]) + resid
};

struct GemmParams {
  const void* A; const void* B; void* C;
  const float* bias; const float* resid; const float* gamma;
  long long sAm, sAk, sAbo, sAbi;    // element strides
  long long sBk, sBn, sBbo, sBbi;
  long long sCm, sCn, sCbo, sCbi;
  int K; int innerB;
};

__device__ __forceinline__ unsigned short f2bf(float f) {
  unsigned int u = __float_as_uint(f);
  u += 0x7FFFu + ((u >> 16) & 1u);     // RTNE
  return (unsigned short)(u >> 16);
}
__device__ __forceinline__ unsigned int pack2bf(float lo, float hi) {
  return (unsigned int)f2bf(lo) | ((unsigned int)f2bf(hi) << 16);
}

union FragAB { v16bf v; unsigned int u[8]; };
union FragC  { v8f   v; float        f[8]; };

template <int ALD, int BLD, int ABF, int BBF, int MODE>
__global__ __launch_bounds__(256)
void gemm_k(GemmParams p) {
  __shared__ unsigned short As[64 * PITCH];   // [m][k]
  __shared__ unsigned short Bs[64 * PITCH];   // [n][k]

  const int tid   = threadIdx.x;
  const int wid   = tid >> 5;
  const int lane  = tid & 31;
  const int l15   = lane & 15;
  const int half  = lane >> 4;
  const int waveM = wid >> 1;           // 0..3
  const int waveN = wid & 1;            // 0..1

  const int zOut = blockIdx.z / p.innerB;
  const int zIn  = blockIdx.z % p.innerB;

  const long long m0 = (long long)blockIdx.x * 64;
  const long long n0 = (long long)blockIdx.y * 64;

  const long long offA = (long long)zOut * p.sAbo + (long long)zIn * p.sAbi + m0 * p.sAm;
  const long long offB = (long long)zOut * p.sBbo + (long long)zIn * p.sBbi + n0 * p.sBn;

  const float*          Af = (const float*)p.A;
  const unsigned short* Ah = (const unsigned short*)p.A;
  const float*          Bf = (const float*)p.B;
  const unsigned short* Bh = (const unsigned short*)p.B;

  // hoisted per-thread loader bases (element offsets)
  const long long aBaseK  = offA + (long long)(tid >> 2) * p.sAm + (tid & 3) * 8;
  const long long aBaseMN = offA + (long long)(tid & 7) * 8 + (long long)(tid >> 3) * p.sAk;
  const long long bBaseK  = offB + (long long)(tid >> 2) * p.sBn + (tid & 3) * 8;
  const long long bBaseMN = offB + (long long)(tid & 7) * 8 + (long long)(tid >> 3) * p.sBk;

  FragC acc0 = {};
  FragC acc1 = {};

  for (int k0 = 0; k0 < p.K; k0 += 32) {
    __syncthreads();

    // ------------------- stage A tile (64m x 32k) -------------------
    if constexpr (ALD == LD_K) {                 // sAk == 1
      const long long off = aBaseK + k0;
      unsigned int w0, w1, w2, w3;
      if constexpr (ABF) {
        const uint4 d = *(const uint4*)(Ah + off);
        w0 = d.x; w1 = d.y; w2 = d.z; w3 = d.w;
      } else {
        const float4 f0 = *(const float4*)(Af + off);
        const float4 f1 = *(const float4*)(Af + off + 4);
        w0 = pack2bf(f0.x, f0.y); w1 = pack2bf(f0.z, f0.w);
        w2 = pack2bf(f1.x, f1.y); w3 = pack2bf(f1.z, f1.w);
      }
      *(uint4*)&As[(tid >> 2) * PITCH + (tid & 3) * 8] = make_uint4(w0, w1, w2, w3);
    } else if constexpr (ALD == LD_MN) {         // sAm == 1
      const long long off = aBaseMN + (long long)k0 * p.sAk;
      const int kk = tid >> 3, mm0 = (tid & 7) * 8;
      unsigned short h[8];
      if constexpr (ABF) {
        const uint4 d = *(const uint4*)(Ah + off);
        h[0] = (unsigned short)d.x; h[1] = (unsigned short)(d.x >> 16);
        h[2] = (unsigned short)d.y; h[3] = (unsigned short)(d.y >> 16);
        h[4] = (unsigned short)d.z; h[5] = (unsigned short)(d.z >> 16);
        h[6] = (unsigned short)d.w; h[7] = (unsigned short)(d.w >> 16);
      } else {
        const float4 f0 = *(const float4*)(Af + off);
        const float4 f1 = *(const float4*)(Af + off + 4);
        h[0] = f2bf(f0.x); h[1] = f2bf(f0.y); h[2] = f2bf(f0.z); h[3] = f2bf(f0.w);
        h[4] = f2bf(f1.x); h[5] = f2bf(f1.y); h[6] = f2bf(f1.z); h[7] = f2bf(f1.w);
      }
      #pragma unroll
      for (int i = 0; i < 8; ++i) As[(mm0 + i) * PITCH + kk] = h[i];
    } else {                                     // scalar fallback
      #pragma unroll
      for (int j = 0; j < 8; ++j) {
        int e = tid + 256 * j;
        int m = e & 63, kk = e >> 6;
        long long off = offA + (long long)m * p.sAm + (long long)(k0 + kk) * p.sAk;
        As[m * PITCH + kk] = ABF ? Ah[off] : f2bf(Af[off]);
      }
    }

    // ------------------- stage B tile (32k x 64n) -> Bs[n][k] -------------------
    if constexpr (BLD == LD_K) {                 // sBk == 1
      const long long off = bBaseK + k0;
      unsigned int w0, w1, w2, w3;
      if constexpr (BBF) {
        const uint4 d = *(const uint4*)(Bh + off);
        w0 = d.x; w1 = d.y; w2 = d.z; w3 = d.w;
      } else {
        const float4 f0 = *(const float4*)(Bf + off);
        const float4 f1 = *(const float4*)(Bf + off + 4);
        w0 = pack2bf(f0.x, f0.y); w1 = pack2bf(f0.z, f0.w);
        w2 = pack2bf(f1.x, f1.y); w3 = pack2bf(f1.z, f1.w);
      }
      *(uint4*)&Bs[(tid >> 2) * PITCH + (tid & 3) * 8] = make_uint4(w0, w1, w2, w3);
    } else if constexpr (BLD == LD_MN) {         // sBn == 1
      const long long off = bBaseMN + (long long)k0 * p.sBk;
      const int kk = tid >> 3, nn0 = (tid & 7) * 8;
      unsigned short h[8];
      if constexpr (BBF) {
        const uint4 d = *(const uint4*)(Bh + off);
        h[0] = (unsigned short)d.x; h[1] = (unsigned short)(d.x >> 16);
        h[2] = (unsigned short)d.y; h[3] = (unsigned short)(d.y >> 16);
        h[4] = (unsigned short)d.z; h[5] = (unsigned short)(d.z >> 16);
        h[6] = (unsigned short)d.w; h[7] = (unsigned short)(d.w >> 16);
      } else {
        const float4 f0 = *(const float4*)(Bf + off);
        const float4 f1 = *(const float4*)(Bf + off + 4);
        h[0] = f2bf(f0.x); h[1] = f2bf(f0.y); h[2] = f2bf(f0.z); h[3] = f2bf(f0.w);
        h[4] = f2bf(f1.x); h[5] = f2bf(f1.y); h[6] = f2bf(f1.z); h[7] = f2bf(f1.w);
      }
      #pragma unroll
      for (int i = 0; i < 8; ++i) Bs[(nn0 + i) * PITCH + kk] = h[i];
    } else {
      #pragma unroll
      for (int j = 0; j < 8; ++j) {
        int e = tid + 256 * j;
        int n = e & 63, kk = e >> 6;
        long long off = offB + (long long)(k0 + kk) * p.sBk + (long long)n * p.sBn;
        Bs[n * PITCH + kk] = BBF ? Bh[off] : f2bf(Bf[off]);
      }
    }
    __syncthreads();

    // ---- gather fragments per CDNA5 WMMA VGPR layouts (CDNA5 ISA 7.12.2) ----
    // A 16x32 bf16: half 0 -> K {0..7,16..23}; half 1 -> K {8..15,24..31}
    FragAB fa;
    const int am = waveM * 16 + l15;
    #pragma unroll
    for (int j = 0; j < 8; ++j) {
      int kb = (j < 4 ? 2 * j : 8 + 2 * j) + 8 * half;
      fa.u[j] = *(const unsigned int*)&As[am * PITCH + kb];
    }
    // B 32x16 bf16: half 0 -> K {0..7,16..23}; half 1 -> K {8..15,24..31}
    FragAB fb0, fb1;
    const int bn0 = waveN * 32 + l15;
    const int bn1 = bn0 + 16;
    #pragma unroll
    for (int j = 0; j < 8; ++j) {
      int kb = (j < 4 ? 2 * j : 8 + 2 * j) + 8 * half;
      fb0.u[j] = *(const unsigned int*)&Bs[bn0 * PITCH + kb];
      fb1.u[j] = *(const unsigned int*)&Bs[bn1 * PITCH + kb];
    }

    acc0.v = __builtin_amdgcn_wmma_f32_16x16x32_bf16(
        false, fa.v, false, fb0.v, (short)0, acc0.v, false, false);
    acc1.v = __builtin_amdgcn_wmma_f32_16x16x32_bf16(
        false, fa.v, false, fb1.v, (short)0, acc1.v, false, false);
  }

  // ---- epilogue: D VGPR r holds M = r + 8*half, N = lane&15 ----
  const long long cBase = (long long)zOut * p.sCbo + (long long)zIn * p.sCbi;
  float*          Cf = (float*)p.C;
  unsigned short* Ch = (unsigned short*)p.C;

  #pragma unroll
  for (int sub = 0; sub < 2; ++sub) {
    FragC a; a.v = (sub == 0) ? acc0.v : acc1.v;
    long long n = n0 + waveN * 32 + sub * 16 + l15;
    #pragma unroll
    for (int r = 0; r < 8; ++r) {
      long long m   = m0 + waveM * 16 + r + 8 * half;
      long long off = cBase + m * p.sCm + n * p.sCn;
      float v = a.f[r];
      if constexpr (MODE == MODE_STORE_F32) {
        Cf[off] = v;
      } else if constexpr (MODE == MODE_ACCUM_F32) {
        Cf[off] += v;
      } else if constexpr (MODE == MODE_STORE_BF16_B) {
        Ch[off] = f2bf(v + p.bias[m]);
      } else {
        float g = p.gamma[0];
        Cf[off] = g * (v + 6.0f * p.bias[m]) + p.resid[off];
      }
    }
  }
}

// ---------------------------------------------------------------------------
// Joint softmax over concat([part1, part2]) per row; one masked element in
// part1 at (row / maskDiv) % maskMod.  One wave32 per row, bf16 outputs.
// ---------------------------------------------------------------------------
template <int C1, int C2>
__global__ __launch_bounds__(256)
void softmax_pair(const float* __restrict__ e1, const float* __restrict__ e2,
                  unsigned short* __restrict__ p1, unsigned short* __restrict__ p2,
                  int maskDiv, int maskMod, int rows) {
  int row = blockIdx.x * 8 + (threadIdx.x >> 5);
  if (row >= rows) return;
  int lane = threadIdx.x & 31;
  int maskIdx = (row / maskDiv) % maskMod;

  float r1[C1], r2[C2];
  float mx = -3.0e38f;
  long long b1 = (long long)row * (C1 * 32);
  long long b2 = (long long)row * (C2 * 32);
  #pragma unroll
  for (int j = 0; j < C1; ++j) {
    int i = lane + 32 * j;
    float v = e1[b1 + i];
    if (i == maskIdx) v = -3.0e38f;
    r1[j] = v; mx = fmaxf(mx, v);
  }
  #pragma unroll
  for (int j = 0; j < C2; ++j) {
    float v = e2[b2 + lane + 32 * j];
    r2[j] = v; mx = fmaxf(mx, v);
  }
  for (int s = 16; s > 0; s >>= 1) mx = fmaxf(mx, __shfl_xor(mx, s, 32));
  float sum = 0.0f;
  #pragma unroll
  for (int j = 0; j < C1; ++j) { r1[j] = __expf(r1[j] - mx); sum += r1[j]; }
  #pragma unroll
  for (int j = 0; j < C2; ++j) { r2[j] = __expf(r2[j] - mx); sum += r2[j]; }
  for (int s = 16; s > 0; s >>= 1) sum += __shfl_xor(sum, s, 32);
  float inv = 1.0f / sum;
  #pragma unroll
  for (int j = 0; j < C1; ++j) p1[b1 + lane + 32 * j] = f2bf(r1[j] * inv);
  #pragma unroll
  for (int j = 0; j < C2; ++j) p2[b2 + lane + 32 * j] = f2bf(r2[j] * inv);
}

// ---------------------------------------------------------------------------
// Host side
// ---------------------------------------------------------------------------
template <int ALD, int BLD, int ABF, int BBF, int MODE>
static void gemm(hipStream_t st, int M, int N, int K, int outer, int inner,
                 const void* A, long long sAm, long long sAk,
                 long long sAbo, long long sAbi,
                 const void* Bp, long long sBk, long long sBn,
                 long long sBbo, long long sBbi,
                 void* Cp, long long sCm, long long sCn,
                 long long sCbo, long long sCbi,
                 const float* bias = nullptr, const float* resid = nullptr,
                 const float* gamma = nullptr) {
  GemmParams p;
  p.A = A; p.B = Bp; p.C = Cp; p.bias = bias; p.resid = resid; p.gamma = gamma;
  p.sAm = sAm; p.sAk = sAk; p.sAbo = sAbo; p.sAbi = sAbi;
  p.sBk = sBk; p.sBn = sBn; p.sBbo = sBbo; p.sBbi = sBbi;
  p.sCm = sCm; p.sCn = sCn; p.sCbo = sCbo; p.sCbi = sCbi;
  p.K = K; p.innerB = inner;
  dim3 g((unsigned)(M / 64), (unsigned)(N / 64), (unsigned)(outer * inner));
  gemm_k<ALD, BLD, ABF, BBF, MODE><<<g, 256, 0, st>>>(p);
}

extern "C" void kernel_launch(void* const* d_in, const int* in_sizes, int n_in,
                              void* d_out, int out_size, void* d_ws, size_t ws_size,
                              hipStream_t stream) {
  (void)in_sizes; (void)n_in; (void)out_size; (void)ws_size;
  const float* x   = (const float*)d_in[0];
  const float* wq  = (const float*)d_in[1];
  const float* bq  = (const float*)d_in[2];
  const float* wk  = (const float*)d_in[3];
  const float* bk  = (const float*)d_in[4];
  const float* wv  = (const float*)d_in[5];
  const float* bv  = (const float*)d_in[6];
  const float* wo  = (const float*)d_in[7];
  const float* bo_ = (const float*)d_in[8];
  const float* gam = (const float*)d_in[9];
  float* out = (float*)d_out;

  const long long Bn = 8, Dd = 512, Cc = 64, Vv = 128, Tt = 128;
  const long long CVT = Cc * Vv * Tt;          // 1048576

  // -------- workspace carve-up (~304 MiB) --------
  char* ws = (char*)d_ws;
  size_t o = 0;
  unsigned short* qb = (unsigned short*)(ws + o); o += (size_t)Bn*CVT*2;
  unsigned short* kb = (unsigned short*)(ws + o); o += (size_t)Bn*CVT*2;
  unsigned short* vb = (unsigned short*)(ws + o); o += (size_t)Bn*CVT*2;
  float* Lsmall = (float*)(ws + o); o += (size_t)Bn*Cc*Vv*Cc*4;
  float* Lbig1  = (float*)(ws + o); o += (size_t)Bn*Cc*Vv*Vv*4;
  float* Lbig2  = (float*)(ws + o); o += (size_t)Bn*Vv*Tt*Tt*4;
  unsigned short* a_vcc = (unsigned short*)(ws + o); o += (size_t)Bn*Cc*Vv*Cc*2;
  unsigned short* a_cvv = (unsigned short*)(ws + o); o += (size_t)Bn*Cc*Vv*Vv*2;
  unsigned short* a_tcc = (unsigned short*)(ws + o); o += (size_t)Bn*Cc*Tt*Cc*2;
  unsigned short* a_ctt = (unsigned short*)(ws + o); o += (size_t)Bn*Cc*Tt*Tt*2;
  unsigned short* a_tvv = (unsigned short*)(ws + o); o += (size_t)Bn*Vv*Tt*Vv*2;
  unsigned short* a_vtt = (unsigned short*)(ws + o); o += (size_t)Bn*Vv*Tt*Tt*2;
  float* total = (float*)(ws + o); o += (size_t)Bn*CVT*4;

  // -------- 1) projections: q,k,v = W @ x + b --------
  gemm<LD_K, LD_MN, 0, 0, MODE_STORE_BF16_B>(stream, 64, 16384, 512, 8, 1,
       wq, 512, 1, 0, 0,  x, Vv*Tt, 1, Dd*Vv*Tt, 0,
       qb, Vv*Tt, 1, CVT, 0, bq);
  gemm<LD_K, LD_MN, 0, 0, MODE_STORE_BF16_B>(stream, 64, 16384, 512, 8, 1,
       wk, 512, 1, 0, 0,  x, Vv*Tt, 1, Dd*Vv*Tt, 0,
       kb, Vv*Tt, 1, CVT, 0, bk);
  gemm<LD_K, LD_MN, 0, 0, MODE_STORE_BF16_B>(stream, 64, 16384, 512, 8, 1,
       wv, 512, 1, 0, 0,  x, Vv*Tt, 1, Dd*Vv*Tt, 0,
       vb, Vv*Tt, 1, CVT, 0, bv);

  // -------- 2) group (c,v):  e_vcc, e_cvv --------
  gemm<LD_K, LD_K, 1, 1, MODE_STORE_F32>(stream, 64, 64, 128, 8, 128,   // (b,v)
       qb, Vv*Tt, 1, CVT, Tt,  kb, 1, Vv*Tt, CVT, Tt,
       Lsmall, Vv*Cc, 1, Cc*Vv*Cc, Cc);
  gemm<LD_K, LD_K, 1, 1, MODE_STORE_F32>(stream, 128, 128, 128, 8, 64,  // (b,c)
       qb, Tt, 1, CVT, Vv*Tt,  kb, 1, Tt, CVT, Vv*Tt,
       Lbig1, Vv, 1, Cc*Vv*Vv, Vv*Vv);
  softmax_pair<2,4><<<65536/8, 256, 0, stream>>>(Lsmall, Lbig1, a_vcc, a_cvv,
                                                 128, 64, 65536);

  // -------- 3) group (c,t):  e_tcc, e_ctt --------
  gemm<LD_NONE, LD_NONE, 1, 1, MODE_STORE_F32>(stream, 64, 64, 128, 8, 128, // (b,t)
       qb, Vv*Tt, Tt, CVT, 1,  kb, Tt, Vv*Tt, CVT, 1,
       Lsmall, Tt*Cc, 1, Cc*Tt*Cc, Cc);
  gemm<LD_MN, LD_MN, 1, 1, MODE_STORE_F32>(stream, 128, 128, 128, 8, 64,    // (b,c)
       qb, 1, Tt, CVT, Vv*Tt,  kb, Tt, 1, CVT, Vv*Tt,
       Lbig1, Tt, 1, Cc*Tt*Tt, Tt*Tt);
  softmax_pair<2,4><<<65536/8, 256, 0, stream>>>(Lsmall, Lbig1, a_tcc, a_ctt,
                                                 128, 64, 65536);

  // -------- 4) group (v,t):  e_tvv, e_vtt --------
  gemm<LD_NONE, LD_NONE, 1, 1, MODE_STORE_F32>(stream, 128, 128, 64, 8, 128, // (b,t)
       qb, Tt, Vv*Tt, CVT, 1,  kb, Vv*Tt, Tt, CVT, 1,
       Lbig1, Tt*Vv, 1, Vv*Tt*Vv, Vv);
  gemm<LD_MN, LD_MN, 1, 1, MODE_STORE_F32>(stream, 128, 128, 64, 8, 128,     // (b,v)
       qb, 1, Vv*Tt, CVT, Tt,  kb, Vv*Tt, 1, CVT, Tt,
       Lbig2, Tt, 1, Vv*Tt*Tt, Tt*Tt);
  softmax_pair<4,4><<<131072/8, 256, 0, stream>>>(Lbig1, Lbig2, a_tvv, a_vtt,
                                                  128, 128, 131072);

  // -------- 5) six attention applications into `total` --------
  // out_vt: val @ a_ctt           (b,c)   first one STOREs (initializes total)
  gemm<LD_K, LD_MN, 1, 1, MODE_STORE_F32>(stream, 128, 128, 128, 8, 64,
       vb, Tt, 1, CVT, Vv*Tt,  a_ctt, Tt, 1, Cc*Tt*Tt, Tt*Tt,
       total, Tt, 1, CVT, Vv*Tt);
  // out_tv: a_cvv^T @ val         (b,c)
  gemm<LD_MN, LD_MN, 1, 1, MODE_ACCUM_F32>(stream, 128, 128, 128, 8, 64,
       a_cvv, 1, Vv, Cc*Vv*Vv, Vv*Vv,  vb, Tt, 1, CVT, Vv*Tt,
       total, Tt, 1, CVT, Vv*Tt);
  // out_ct: val @ a_vtt           (b,v)
  gemm<LD_K, LD_MN, 1, 1, MODE_ACCUM_F32>(stream, 64, 128, 128, 8, 128,
       vb, Vv*Tt, 1, CVT, Tt,  a_vtt, Tt, 1, Vv*Tt*Tt, Tt*Tt,
       total, Vv*Tt, 1, CVT, Tt);
  // out_tc: a_vcc^T @ val         (b,v)
  gemm<LD_MN, LD_MN, 1, 1, MODE_ACCUM_F32>(stream, 64, 128, 64, 8, 128,
       a_vcc, 1, Vv*Cc, Cc*Vv*Cc, Cc,  vb, Vv*Tt, 1, CVT, Tt,
       total, Vv*Tt, 1, CVT, Tt);
  // out_cv: val @ a_tvv           (b,t)
  gemm<LD_NONE, LD_MN, 1, 1, MODE_ACCUM_F32>(stream, 64, 128, 128, 8, 128,
       vb, Vv*Tt, Tt, CVT, 1,  a_tvv, Tt*Vv, 1, Vv*Tt*Vv, Vv,
       total, Vv*Tt, Tt, CVT, 1);
  // out_vc: a_tcc^T @ val         (b,t)
  gemm<LD_MN, LD_NONE, 1, 1, MODE_ACCUM_F32>(stream, 64, 128, 64, 8, 128,
       a_tcc, 1, Tt*Cc, Cc*Tt*Cc, Cc,  vb, Vv*Tt, Tt, CVT, 1,
       total, Vv*Tt, Tt, CVT, 1);

  // -------- 6) final: out = gamma*(wo @ total + 6*bo) + x --------
  gemm<LD_K, LD_MN, 0, 0, MODE_FINAL>(stream, 512, 16384, 64, 8, 1,
       wo, 64, 1, 0, 0,  total, Vv*Tt, 1, CVT, 0,
       out, Vv*Tt, 1, Dd*Vv*Tt, 0, bo_, x, gam);
}
